// Tprod_13417477833487
// MI455X (gfx1250) — matrix operand order repaired and checked by reading, not compiled, not run on hardware
//
#include <hip/hip_runtime.h>

typedef __bf16 bf16;
typedef __attribute__((ext_vector_type(16))) __bf16 v16bf;
typedef __attribute__((ext_vector_type(8)))  __bf16 v8bf;
typedef __attribute__((ext_vector_type(4)))  __bf16 v4bf;
typedef __attribute__((ext_vector_type(2)))  __bf16 v2bf;
typedef __attribute__((ext_vector_type(8)))  float  v8f;

#define BB 8
#define NN 256
#define MM 128
#define DD 128

// ---------- WMMA helpers (CDNA5 wave32, 16x16x32 bf16 -> f32) ----------

// A-matrix 16x32 (MxK), 16-bit: lane L holds row M=L&15.
// lanes 0-15:  VGPR0-3 = K 0..7,  VGPR4-7 = K 16..23
// lanes 16-31: VGPR0-3 = K 8..15, VGPR4-7 = K 24..31
__device__ __forceinline__ v16bf frag_a(const bf16* row, int lane) {
  const int koff = (lane & 16) ? 8 : 0;
  v16bf r;
  ((v8bf*)&r)[0] = *(const v8bf*)(row + koff);
  ((v8bf*)&r)[1] = *(const v8bf*)(row + koff + 16);
  return r;
}

// B-matrix 32x16 (KxN), 16-bit: lane L holds column N=L&15.
// lanes 0-15: K 0..7, K 16..23; lanes 16-31: K 8..15, K 24..31 (matches frag_a).
__device__ __forceinline__ v16bf frag_b(const bf16* col, int lane) {
  const int koff = (lane & 16) ? 8 : 0;
  v16bf r;
  ((v8bf*)&r)[0] = *(const v8bf*)(col + koff);
  ((v8bf*)&r)[1] = *(const v8bf*)(col + koff + 16);
  return r;
}

__device__ __forceinline__ v8f wmma_bf16(v16bf a, v16bf b, v8f c) {
  // (neg_a, A, neg_b, B, c_mod, C, reuse_a, reuse_b)
  return __builtin_amdgcn_wmma_f32_16x16x32_bf16(false, a, false, b, (short)0, c,
                                                 false, false);
}

__device__ __forceinline__ v8f vzero() {
  v8f z = {0.f, 0.f, 0.f, 0.f, 0.f, 0.f, 0.f, 0.f};
  return z;
}

__device__ __forceinline__ bf16 tobf(float x) { return (bf16)x; }

__device__ __forceinline__ v4bf pk4(float a, float b, float c, float d) {
  v4bf r;
  r[0] = tobf(a); r[1] = tobf(b); r[2] = tobf(c); r[3] = tobf(d);
  return r;
}
__device__ __forceinline__ v2bf pk2(float a, float b) {
  v2bf r;
  r[0] = tobf(a); r[1] = tobf(b);
  return r;
}

// ---------- Kernel 1: s_v2t[b,d] = sum_m gv2[b,m,m,d] ----------
__global__ void k_trace(const float* __restrict__ gv2, float* __restrict__ sv2t) {
  const int b = blockIdx.x;
  const int d = threadIdx.x;
  float acc = 0.f;
  for (int m = 0; m < MM; ++m)
    acc += gv2[(((size_t)b * MM + m) * MM + m) * DD + d];
  sv2t[b * DD + d] = acc;
}

// ---------- Kernel 2: v_new, fully fused ----------
// One workgroup per (b, n-tile 16, c-tile 16). 512 threads = 16 waves.
// Phase 1: T[n,c,d] = sum_m v[b,n,m,d]*gv2[b,m,c,d]; also captures X3 = gs*v tile
//          while v streams through (no phase-2 re-read of v).
// Phase 2: v_new = ( [T | s*gv | gs*v] @ [W1;W2;W3]^T ) / 3 as one K=384 GEMM
//          over three flat K=128 segment buffers.
// LDS: phase1 A(128K)+B(128K)+X3(64K) = 327680 B (full 320KB WGP LDS).
//      phase2 overlays X1,X2 on A; Wc on B; X3 persists.
__global__ __launch_bounds__(512) void k_vnew(
    const float* __restrict__ s,  const float* __restrict__ v,
    const float* __restrict__ gs, const float* __restrict__ gv,
    const float* __restrict__ gv2,
    const float* __restrict__ w1, const float* __restrict__ w2,
    const float* __restrict__ w3, float* __restrict__ vnew) {
  extern __shared__ unsigned char smem[];
  bf16* A_ls = (bf16*)smem;                  // [128 d][16 n][32 m]
  bf16* B_ls = (bf16*)(smem + 131072);       // [128 d][16 c][32 m]
  bf16* X1   = (bf16*)smem;                  // [256 p][128 k]  (overlays A_ls)
  bf16* X2   = (bf16*)(smem + 65536);        // [256 p][128 k]  (overlays A_ls)
  bf16* Wc   = (bf16*)(smem + 131072);       // [128 e][384 k]  (overlays B_ls)
  bf16* X3   = (bf16*)(smem + 262144);       // [256 p][128 k]  (persistent)

  const int tid  = threadIdx.x;
  const int lane = tid & 31;
  const int wave = tid >> 5;
  const int blk  = blockIdx.x;
  const int c0 = (blk & 7) * 16;
  const int n0 = ((blk >> 3) & 15) * 16;
  const int b  = blk >> 7;

  v8f acc[8];
#pragma unroll
  for (int ch = 0; ch < 8; ++ch) acc[ch] = vzero();

  // ---- Phase 1: channel-batched GEMM over m (4 chunks of K=32) ----
  for (int mc = 0; mc < 4; ++mc) {
    const int m0 = mc * 32;
    // 16 rc * 8 m-groups * 32 d4 = 4096 iterations; each handles 4m x 4d.
    for (int i = tid; i < 4096; i += 512) {
      const int mg = i & 7;          // m-group (4 m's)
      const int d4 = (i >> 3) & 31;  // float4 index along d
      const int rc = i >> 8;         // n (for A) / c (for B) within tile
      const int m  = mg * 4;
      const int db = d4 * 4;
      const float* pa = v   + (((size_t)b * NN + n0 + rc) * MM + m0 + m) * DD + db;
      const float* pb = gv2 + (((size_t)b * MM + m0 + m) * MM + c0 + rc) * DD + db;
      const float4 a0 = *(const float4*)(pa);
      const float4 a1 = *(const float4*)(pa + DD);
      const float4 a2 = *(const float4*)(pa + 2 * DD);
      const float4 a3 = *(const float4*)(pa + 3 * DD);
      const float4 b0 = *(const float4*)(pb);
      const float4 b1 = *(const float4*)(pb + (size_t)MM * DD);
      const float4 b2 = *(const float4*)(pb + 2 * (size_t)MM * DD);
      const float4 b3 = *(const float4*)(pb + 3 * (size_t)MM * DD);
      // A/B tiles: pack 4 m-adjacent bf16 -> one b64 store per d
      *(v4bf*)&A_ls[((db + 0) * 16 + rc) * 32 + m] = pk4(a0.x, a1.x, a2.x, a3.x);
      *(v4bf*)&A_ls[((db + 1) * 16 + rc) * 32 + m] = pk4(a0.y, a1.y, a2.y, a3.y);
      *(v4bf*)&A_ls[((db + 2) * 16 + rc) * 32 + m] = pk4(a0.z, a1.z, a2.z, a3.z);
      *(v4bf*)&A_ls[((db + 3) * 16 + rc) * 32 + m] = pk4(a0.w, a1.w, a2.w, a3.w);
      *(v4bf*)&B_ls[((db + 0) * 16 + rc) * 32 + m] = pk4(b0.x, b1.x, b2.x, b3.x);
      *(v4bf*)&B_ls[((db + 1) * 16 + rc) * 32 + m] = pk4(b0.y, b1.y, b2.y, b3.y);
      *(v4bf*)&B_ls[((db + 2) * 16 + rc) * 32 + m] = pk4(b0.z, b1.z, b2.z, b3.z);
      *(v4bf*)&B_ls[((db + 3) * 16 + rc) * 32 + m] = pk4(b0.w, b1.w, b2.w, b3.w);
      // Capture X3 = gs * v while the v tile streams through (m-group 4-aligned,
      // c-tile 16-aligned => group is entirely inside or outside the c-range).
      if ((unsigned)(m0 + m - c0) < 16u) {
        const int c = m0 + m - c0;
        const float4 g4 = *(const float4*)(gs + b * DD + db);
        *(v4bf*)&X3[(rc * 16 + c + 0) * 128 + db] =
            pk4(a0.x * g4.x, a0.y * g4.y, a0.z * g4.z, a0.w * g4.w);
        *(v4bf*)&X3[(rc * 16 + c + 1) * 128 + db] =
            pk4(a1.x * g4.x, a1.y * g4.y, a1.z * g4.z, a1.w * g4.w);
        *(v4bf*)&X3[(rc * 16 + c + 2) * 128 + db] =
            pk4(a2.x * g4.x, a2.y * g4.y, a2.z * g4.z, a2.w * g4.w);
        *(v4bf*)&X3[(rc * 16 + c + 3) * 128 + db] =
            pk4(a3.x * g4.x, a3.y * g4.y, a3.z * g4.z, a3.w * g4.w);
      }
    }
    __syncthreads();
#pragma unroll
    for (int ch = 0; ch < 8; ++ch) {
      const int dch = wave * 8 + ch;
      v16bf fa = frag_a(&A_ls[(dch * 16 + (lane & 15)) * 32], lane);
      v16bf fb = frag_b(&B_ls[(dch * 16 + (lane & 15)) * 32], lane);
      acc[ch] = wmma_bf16(fa, fb, acc[ch]);
    }
    __syncthreads();  // protect LDS before next stage / phase-2 overwrite
  }

  // ---- Phase 2: spill T -> X1, build X2 = s*gv, stage stacked weights ----
#pragma unroll
  for (int ch = 0; ch < 8; ch += 2) {
    const int dch = wave * 8 + ch;  // even -> 4B-aligned paired store
#pragma unroll
    for (int j = 0; j < 8; ++j) {
      const int n = j + ((lane & 16) ? 8 : 0);
      const int c = lane & 15;
      *(v2bf*)&X1[(n * 16 + c) * 128 + dch] = pk2(acc[ch][j], acc[ch + 1][j]);
    }
  }
  for (int i = tid; i < 256 * 64; i += 512) {
    const int dd = (i & 63) * 2;
    const int p  = i >> 6;
    const int n = p >> 4, c = p & 15;
    const float2 sv = *(const float2*)(s + ((size_t)b * NN + n0 + n) * DD + dd);
    const float2 gg = *(const float2*)(gv + ((size_t)b * MM + c0 + c) * DD + dd);
    *(v2bf*)&X2[p * 128 + dd] = pk2(sv.x * gg.x, sv.y * gg.y);
  }
  for (int i = tid; i < DD * 64; i += 512) {
    const int k = (i & 63) * 2;
    const int e = i >> 6;
    const float2 q1 = *(const float2*)(w1 + (size_t)e * DD + k);
    const float2 q2 = *(const float2*)(w2 + (size_t)e * DD + k);
    const float2 q3 = *(const float2*)(w3 + (size_t)e * DD + k);
    *(v2bf*)&Wc[e * 384 + k]       = pk2(q1.x, q1.y);
    *(v2bf*)&Wc[e * 384 + 128 + k] = pk2(q2.x, q2.y);
    *(v2bf*)&Wc[e * 384 + 256 + k] = pk2(q3.x, q3.y);
  }
  __syncthreads();

  // GEMM: wave w computes rows p = w*16..w*16+15, all 128 output cols.
  // K = 384 over three flat segments {X1, X2, X3} matched to {W1, W2, W3}.
  v8f acc2[8];
#pragma unroll
  for (int t = 0; t < 8; ++t) acc2[t] = vzero();
  const int prow = wave * 16 + (lane & 15);
  const bf16* xseg[3] = {X1, X2, X3};
#pragma unroll
  for (int seg = 0; seg < 3; ++seg) {
    const bf16* Xb = xseg[seg];
#pragma unroll
    for (int kq = 0; kq < 4; ++kq) {
      v16bf fa = frag_a(&Xb[prow * 128 + kq * 32], lane);
#pragma unroll
      for (int ctl = 0; ctl < 8; ++ctl) {
        const int e = ctl * 16 + (lane & 15);
        v16bf fb = frag_b(&Wc[e * 384 + seg * 128 + kq * 32], lane);
        acc2[ctl] = wmma_bf16(fa, fb, acc2[ctl]);
      }
    }
  }
  const float inv3 = 1.0f / 3.0f;
#pragma unroll
  for (int ctl = 0; ctl < 8; ++ctl) {
#pragma unroll
    for (int j = 0; j < 8; ++j) {
      const int Mr = j + ((lane & 16) ? 8 : 0);
      const int p = wave * 16 + Mr;
      const int n = p >> 4, c = p & 15;
      const int e = ctl * 16 + (lane & 15);
      vnew[(((size_t)b * NN + n0 + n) * MM + c0 + c) * DD + e] = acc2[ctl][j] * inv3;
    }
  }
}

// ---------- Kernel 3: s_vv[b,n,d] = sum_m gv[b,m,d] * v_new[b,n,m,d] ----------
// 4 rows per block, float4 per lane.
__global__ void k_svv(const float* __restrict__ gv, const float* __restrict__ vnew,
                      float* __restrict__ svv) {
  const int tid = threadIdx.x;
  const int row = blockIdx.x * 4 + (tid >> 5);  // bn index
  const int b   = row >> 8;                     // N = 256
  const int d4  = (tid & 31) * 4;
  const float* vn = vnew + (size_t)row * MM * DD + d4;
  const float* g  = gv + (size_t)b * MM * DD + d4;
  float4 acc = {0.f, 0.f, 0.f, 0.f};
  for (int m = 0; m < MM; ++m) {
    const float4 gm = *(const float4*)(g + (size_t)m * DD);
    const float4 vm = *(const float4*)(vn + (size_t)m * DD);
    acc.x += gm.x * vm.x;
    acc.y += gm.y * vm.y;
    acc.z += gm.z * vm.z;
    acc.w += gm.w * vm.w;
  }
  *(float4*)(svv + (size_t)row * DD + d4) = acc;
}

// ---------- Kernel 4: fused s-path (gating + MLP), WMMA per 16-row tile ----------
__global__ __launch_bounds__(128) void k_s(
    const float* __restrict__ s,      const float* __restrict__ svv,
    const float* __restrict__ sv2t,   const float* __restrict__ gs,
    const float* __restrict__ lin1_w, const float* __restrict__ lin1_b,
    const float* __restrict__ lin2_w, const float* __restrict__ lin2_b,
    const float* __restrict__ lin3_w, const float* __restrict__ lin3_b,
    const float* __restrict__ mw1,    const float* __restrict__ mb1,
    const float* __restrict__ mw2,    const float* __restrict__ mb2,
    float* __restrict__ out) {
  extern __shared__ unsigned char smem[];
  bf16* W2l  = (bf16*)smem;             // lin2_w bf16, [e][k]
  bf16* W3l  = (bf16*)(smem + 32768);   // lin3_w
  bf16* Wm1  = (bf16*)(smem + 65536);   // mlp_w1
  bf16* Wm2  = (bf16*)(smem + 98304);   // mlp_w2
  bf16* s_t  = (bf16*)(smem + 131072);  // 16x128 s rows
  bf16* q_t  = (bf16*)(smem + 135168);  // 16x128 s_vv rows
  bf16* midt = (bf16*)(smem + 139264);  // 16x128 gated product
  bf16* h_t  = (bf16*)(smem + 143360);  // 16x128 relu hidden
  float* t1g = (float*)(smem + 147456); // 128 gate values

  const int tid  = threadIdx.x;
  const int lane = tid & 31;
  const int wave = tid >> 5;
  const size_t g0 = (size_t)blockIdx.x * 16;  // first global row (b*N+n)
  const int b = (int)(g0 >> 8);               // 256 rows per batch

  for (int i = tid; i < DD * 64; i += 128) {
    const int k = (i & 63) * 2;
    const int e = i >> 6;
    const size_t o = (size_t)e * DD + k;
    const float2 q2 = *(const float2*)(lin2_w + o);
    const float2 q3 = *(const float2*)(lin3_w + o);
    const float2 qa = *(const float2*)(mw1 + o);
    const float2 qb = *(const float2*)(mw2 + o);
    *(v2bf*)&W2l[o] = pk2(q2.x, q2.y);
    *(v2bf*)&W3l[o] = pk2(q3.x, q3.y);
    *(v2bf*)&Wm1[o] = pk2(qa.x, qa.y);
    *(v2bf*)&Wm2[o] = pk2(qb.x, qb.y);
  }
  for (int i = tid; i < 16 * 64; i += 128) {
    const int idx = i * 2;
    const float2 sv = *(const float2*)(s + g0 * DD + idx);
    const float2 qv = *(const float2*)(svv + g0 * DD + idx);
    *(v2bf*)&s_t[idx] = pk2(sv.x, sv.y);
    *(v2bf*)&q_t[idx] = pk2(qv.x, qv.y);
  }
  {  // t1g[e] = (lin1(s_v2t) + b1)[e] * gs[b,e]   (tiny: plain VALU)
    const int e = tid;
    float a = lin1_b[e];
    for (int k = 0; k < DD; ++k) a += sv2t[b * DD + k] * lin1_w[e * DD + k];
    t1g[e] = a * gs[b * DD + e];
  }
  __syncthreads();

  const int r = lane & 15;
  // GEMM1: t2 = s @ lin2^T, t3 = s_vv @ lin3^T; each wave owns 2 column tiles.
  v8f a2[2], a3[2];
  a2[0] = a2[1] = a3[0] = a3[1] = vzero();
#pragma unroll
  for (int ci = 0; ci < 2; ++ci) {
    const int e0 = (wave * 2 + ci) * 16;
#pragma unroll
    for (int kt = 0; kt < 4; ++kt) {
      v16bf fa  = frag_a(&s_t[r * DD + kt * 32], lane);
      v16bf fb  = frag_b(&W2l[(e0 + r) * DD + kt * 32], lane);
      a2[ci] = wmma_bf16(fa, fb, a2[ci]);
      v16bf fa3 = frag_a(&q_t[r * DD + kt * 32], lane);
      v16bf fb3 = frag_b(&W3l[(e0 + r) * DD + kt * 32], lane);
      a3[ci] = wmma_bf16(fa3, fb3, a3[ci]);
    }
  }
#pragma unroll
  for (int ci = 0; ci < 2; ++ci) {
    const int e = (wave * 2 + ci) * 16 + r;
#pragma unroll
    for (int j = 0; j < 8; ++j) {
      const int Mr = j + ((lane & 16) ? 8 : 0);
      const float t2v = a2[ci][j] + lin2_b[e];
      const float t3v = a3[ci][j] + lin3_b[e];
      midt[Mr * DD + e] = tobf(t1g[e] * t2v * t3v);
    }
  }
  __syncthreads();

  // GEMM2: h = relu(mid @ mlp_w1^T + b)
  v8f ah[2];
  ah[0] = ah[1] = vzero();
#pragma unroll
  for (int ci = 0; ci < 2; ++ci) {
    const int e0 = (wave * 2 + ci) * 16;
#pragma unroll
    for (int kt = 0; kt < 4; ++kt) {
      v16bf fa = frag_a(&midt[r * DD + kt * 32], lane);
      v16bf fb = frag_b(&Wm1[(e0 + r) * DD + kt * 32], lane);
      ah[ci] = wmma_bf16(fa, fb, ah[ci]);
    }
  }
#pragma unroll
  for (int ci = 0; ci < 2; ++ci) {
    const int e = (wave * 2 + ci) * 16 + r;
#pragma unroll
    for (int j = 0; j < 8; ++j) {
      const int Mr = j + ((lane & 16) ? 8 : 0);
      const float hv = ah[ci][j] + mb1[e];
      h_t[Mr * DD + e] = tobf(hv > 0.f ? hv : 0.f);
    }
  }
  __syncthreads();

  // GEMM3: s_new = h @ mlp_w2^T + b  -> global
  v8f ao[2];
  ao[0] = ao[1] = vzero();
#pragma unroll
  for (int ci = 0; ci < 2; ++ci) {
    const int e0 = (wave * 2 + ci) * 16;
#pragma unroll
    for (int kt = 0; kt < 4; ++kt) {
      v16bf fa = frag_a(&h_t[r * DD + kt * 32], lane);
      v16bf fb = frag_b(&Wm2[(e0 + r) * DD + kt * 32], lane);
      ao[ci] = wmma_bf16(fa, fb, ao[ci]);
    }
  }
#pragma unroll
  for (int ci = 0; ci < 2; ++ci) {
    const int e = (wave * 2 + ci) * 16 + r;
#pragma unroll
    for (int j = 0; j < 8; ++j) {
      const int Mr = j + ((lane & 16) ? 8 : 0);
      out[(g0 + Mr) * DD + e] = ao[ci][j] + mb2[e];
    }
  }
}

// ---------- host launch ----------
extern "C" void kernel_launch(void* const* d_in, const int* in_sizes, int n_in,
                              void* d_out, int out_size, void* d_ws, size_t ws_size,
                              hipStream_t stream) {
  (void)in_sizes; (void)n_in; (void)out_size; (void)ws_size;
  const float* s      = (const float*)d_in[0];
  const float* v      = (const float*)d_in[1];
  const float* gs     = (const float*)d_in[2];
  const float* gv     = (const float*)d_in[3];
  const float* gv2    = (const float*)d_in[4];
  const float* lin1_w = (const float*)d_in[5];
  const float* lin1_b = (const float*)d_in[6];
  const float* lin2_w = (const float*)d_in[7];
  const float* lin2_b = (const float*)d_in[8];
  const float* lin3_w = (const float*)d_in[9];
  const float* lin3_b = (const float*)d_in[10];
  const float* w1     = (const float*)d_in[11];  // linv1_w
  const float* w2     = (const float*)d_in[12];  // linv2_w
  const float* w3     = (const float*)d_in[13];  // linv3_w
  const float* mw1    = (const float*)d_in[14];
  const float* mb1    = (const float*)d_in[15];
  const float* mw2    = (const float*)d_in[16];
  const float* mb2    = (const float*)d_in[17];

  float* out_s = (float*)d_out;                   // (B,N,d)
  float* out_v = out_s + (size_t)BB * NN * DD;    // (B,N,M,d)

  float* svv  = (float*)d_ws;                     // B*N*d fp32 (1 MB)
  float* sv2t = svv + (size_t)BB * NN * DD;       // B*d fp32

  k_trace<<<BB, DD, 0, stream>>>(gv2, sv2t);
  k_vnew<<<BB * (NN / 16) * (MM / 16), 512, 327680, stream>>>(
      s, v, gs, gv, gv2, w1, w2, w3, out_v);
  k_svv<<<(BB * NN) / 4, 128, 0, stream>>>(gv, out_v, svv);
  k_s<<<(BB * NN) / 16, 128, 147968, stream>>>(
      s, svv, sv2t, gs, lin1_w, lin1_b, lin2_w, lin2_b, lin3_w, lin3_b,
      mw1, mb1, mw2, mb2, out_s);
}
